// CausalSimpleKT_38611755991719
// MI455X (gfx1250) — matrix operand order repaired and checked by reading, not compiled, not run on hardware
//
#include <hip/hip_runtime.h>
#include <hip/hip_bf16.h>

// ---------------------------------------------------------------------------
// CausalSimpleKT on MI455X (gfx1250, wave32, WMMA)
//
// Decomposition:
//   pre[b,t,:]  = b_su + W_kc@kc_e + W_q@q_e + W_r@r_e      (batch-parallel WMMA GEMM)
//   pre_sk[b,t] = b_sk + <kc_e, W_sk[256:512]>              (parallel dots)
//   pre_sq[b,t] = b_sq + <q_e,  W_sq[256:1024]>
//   recurrence: u = pre[b,t] + W_s @ state ; state = tanh(LN(u))
//               pred = a*sig(<s,Wsk_s>+pre_sk) + (1-a)*sig(<s,Wsq_s>+pre_sq)
// ---------------------------------------------------------------------------

#define B_   256
#define T_   500
#define BT_  (B_ * T_)      // 128000
#define SD_  256
#define KD_  256
#define QD_  768
#define KTOT 1536           // SD+KD+QD+KD (W_su inner dim)
#define KA   1280           // KD+QD+KD    (precomputable inner dim)
#define LDA  1288           // padded LDS row stride (halves): 644 dwords, %64==4
#define MROWS 32            // M rows per workgroup in the precompute GEMM

typedef __attribute__((ext_vector_type(16))) _Float16 v16h;
typedef __attribute__((ext_vector_type(8)))  _Float16 v8h;
typedef __attribute__((ext_vector_type(8)))  float    v8f;

__device__ __forceinline__ v16h combine16(v8h lo, v8h hi) {
    v16h r;
#pragma unroll
    for (int i = 0; i < 8; ++i) { r[i] = lo[i]; r[i + 8] = hi[i]; }
    return r;
}

__device__ __forceinline__ v16h load_a_frag(const _Float16* Arow, int ko) {
    v8h lo = *(const v8h*)(Arow + ko);
    v8h hi = *(const v8h*)(Arow + ko + 8);
    return combine16(lo, hi);
}

__device__ __forceinline__ v8f wmma_f16(v16h a, v16h b, v8f c) {
    return __builtin_amdgcn_wmma_f32_16x16x32_f16(false, a, false, b,
                                                  (short)0, c, false, false);
}

__device__ __forceinline__ float sigmoidf_(float x) {
    return 1.0f / (1.0f + __expf(-x));
}

// ---------------------------------------------------------------------------
// Kernel 0: convert W_su (f32 [256][1536], row = output n, col = input k)
// to f16 with identical layout; B-matrix fragments are read from it directly.
// ---------------------------------------------------------------------------
__global__ __launch_bounds__(256) void prep_w16(const float* __restrict__ W_su,
                                                _Float16* __restrict__ Wf16) {
    int i = blockIdx.x * 256 + threadIdx.x;   // grid = 256*1536/256
    Wf16[i] = (_Float16)W_su[i];
}

// ---------------------------------------------------------------------------
// Kernel 1: pre[m][n] = b_su[n] + sum_k A[m][k] * W_su[n][256+k], m = b*T+t
// 32-row M block per workgroup (two 16-row WMMA tiles) so every B fragment
// feeds 2 WMMAs; register double-buffering of next-k fragments gives the
// scheduler a 4-WMMA latency-hiding window. Dynamic LDS: 32*1288 halves.
// ---------------------------------------------------------------------------
__global__ __launch_bounds__(256) void precompute_gemm(
    const int* __restrict__ qids, const int* __restrict__ kcids,
    const int* __restrict__ resp,
    const float* __restrict__ q_table, const float* __restrict__ kc_table,
    const float* __restrict__ resp_table,
    const _Float16* __restrict__ Wf16, const float* __restrict__ b_su,
    float* __restrict__ pre) {
    extern __shared__ __attribute__((aligned(16))) _Float16 A[]; // [MROWS*LDA]

    const int tid   = threadIdx.x;
    const int mbase = blockIdx.x * MROWS;

    // ---- stage A block: 32 rows x 1280 halves; 8 threads/row, 160 cols each
    {
        const int row = tid >> 3;          // 0..31
        const int seg = tid & 7;           // 0..7
        const long m  = mbase + row;
        const float* qrow = q_table    + (long)qids[m]  * QD_;
        const float* krow = kc_table   + (long)kcids[m] * KD_;
        const float* rrow = resp_table + (long)resp[m]  * KD_;
        for (int j = 0; j < 160; ++j) {
            int col = seg * 160 + j;
            float v;
            if (col < 256)       v = krow[col];
            else if (col < 1024) v = qrow[col - 256];
            else                 v = rrow[col - 1024];
            A[row * LDA + col] = (_Float16)v;
        }
    }
    __syncthreads();

    const int wave  = tid >> 5;
    const int lane  = tid & 31;
    const int l15   = lane & 15;
    const int khalf = (lane < 16) ? 0 : 16;   // A-fragment K sub-offset
    const int kbb   = (lane < 16) ? 0 : 16;   // B-fragment K sub-offset

    const int n0 = wave * 32 + l15;           // N tile 0 column of this lane
    const int n1 = n0 + 16;                   // N tile 1 column
    const _Float16* Bp0 = Wf16 + (long)n0 * KTOT + SD_ + kbb;  // skip state block
    const _Float16* Bp1 = Wf16 + (long)n1 * KTOT + SD_ + kbb;
    const _Float16* Arow0 = &A[l15 * LDA];          // M rows 0..15
    const _Float16* Arow1 = &A[(16 + l15) * LDA];   // M rows 16..31

    v8f acc00 = {}, acc01 = {}, acc10 = {}, acc11 = {};

    v16h a0 = load_a_frag(Arow0, khalf);
    v16h a1 = load_a_frag(Arow1, khalf);
    v16h b0 = *(const v16h*)(Bp0);
    v16h b1 = *(const v16h*)(Bp1);
    for (int kk = 0; kk < KA / 32; ++kk) {
        v16h a0n = a0, a1n = a1, b0n = b0, b1n = b1;
        if (kk + 1 < KA / 32) {               // issue next-k loads first
            const int ko = (kk + 1) * 32 + khalf;
            a0n = load_a_frag(Arow0, ko);
            a1n = load_a_frag(Arow1, ko);
            b0n = *(const v16h*)(Bp0 + (kk + 1) * 32);
            b1n = *(const v16h*)(Bp1 + (kk + 1) * 32);
        }
        acc00 = wmma_f16(a0, b0, acc00);
        acc01 = wmma_f16(a0, b1, acc01);
        acc10 = wmma_f16(a1, b0, acc10);
        acc11 = wmma_f16(a1, b1, acc11);
        a0 = a0n; a1 = a1n; b0 = b0n; b1 = b1n;
    }

    // ---- store C tiles (+bias): VGPR v -> M = v + (lane<16?0:8), N = lane&15
    const int mrow = (lane < 16) ? 0 : 8;
    const int nb0  = wave * 32;
    const int nb1  = nb0 + 16;
    const float bs0 = b_su[nb0 + l15];
    const float bs1 = b_su[nb1 + l15];
#pragma unroll
    for (int v = 0; v < 8; ++v) {
        const long mm0 = mbase + mrow + v;        // M tile 0
        const long mm1 = mm0 + 16;                // M tile 1
        pre[mm0 * 256 + nb0 + l15] = acc00[v] + bs0;
        pre[mm0 * 256 + nb1 + l15] = acc01[v] + bs1;
        pre[mm1 * 256 + nb0 + l15] = acc10[v] + bs0;
        pre[mm1 * 256 + nb1 + l15] = acc11[v] + bs1;
    }
}

// ---------------------------------------------------------------------------
// Kernel 2: pre_sk / pre_sq gate scalars (1 wave per (b,t) row).
// ---------------------------------------------------------------------------
__global__ __launch_bounds__(256) void precompute_scalars(
    const int* __restrict__ qids, const int* __restrict__ kcids,
    const float* __restrict__ q_table, const float* __restrict__ kc_table,
    const float* __restrict__ W_sk, const float* __restrict__ b_sk,
    const float* __restrict__ W_sq, const float* __restrict__ b_sq,
    float* __restrict__ pre_sk, float* __restrict__ pre_sq) {
    const int wave = threadIdx.x >> 5;
    const int lane = threadIdx.x & 31;
    const long idx = (long)blockIdx.x * 8 + wave;   // grid = BT/8

    const float* krow = kc_table + (long)kcids[idx] * KD_;
    const float* qrow = q_table  + (long)qids[idx]  * QD_;
    const float* wk = W_sk + SD_;   // kc half of W_sk
    const float* wq = W_sq + SD_;   // q  half of W_sq

    float dk = 0.f, dq = 0.f;
    for (int c = lane; c < KD_; c += 32) dk += krow[c] * wk[c];
    for (int c = lane; c < QD_; c += 32) dq += qrow[c] * wq[c];
#pragma unroll
    for (int off = 16; off; off >>= 1) {
        dk += __shfl_xor(dk, off, 32);
        dq += __shfl_xor(dq, off, 32);
    }
    if (lane == 0) {
        pre_sk[idx] = dk + b_sk[0];
        pre_sq[idx] = dq + b_sq[0];
    }
}

// ---------------------------------------------------------------------------
// Kernel 3: recurrence. 16 workgroups x 256 threads; WG g owns batch rows
// [16g, 16g+16) for all T steps; state lives in LDS (f32 + f16 mirror).
// W_s B-fragments are loop-invariant over t -> hoisted into registers once
// (16 x v16h per wave); the serial inner loop is then 8 LDS A-load pairs +
// 16 WMMAs + shuffle-reduced gates/LayerNorm, with `pre` streamed+prefetched.
// ---------------------------------------------------------------------------
__global__ __launch_bounds__(256) void recurrence(
    const _Float16* __restrict__ Wf16, const float* __restrict__ pre,
    const float* __restrict__ pre_sk, const float* __restrict__ pre_sq,
    const float* __restrict__ W_sk, const float* __restrict__ W_sq,
    const float* __restrict__ ln_g, const float* __restrict__ ln_b,
    const float* __restrict__ init_state, const float* __restrict__ alpha_logit,
    float* __restrict__ out_preds, float* __restrict__ out_hs) {
    __shared__ __attribute__((aligned(16))) _Float16 s16[16 * 264]; // state f16
    __shared__ __attribute__((aligned(16))) float    sfl[16 * 256]; // state f32
    __shared__ __attribute__((aligned(16))) float    su [16 * 260]; // u scratch

    const int tid = threadIdx.x;
    const int b0  = blockIdx.x * 16;
    const float alpha = sigmoidf_(alpha_logit[0]);

    for (int i = tid; i < 16 * 256; i += 256) {
        const int r = i >> 8, c = i & 255;
        const float v = init_state[c];
        sfl[r * 256 + c] = v;
        s16[r * 264 + c] = (_Float16)v;
    }
    __syncthreads();

    const int wave  = tid >> 5;
    const int lane  = tid & 31;
    const int l15   = lane & 15;
    const int khalf = (lane < 16) ? 0 : 16;
    const int kbb   = (lane < 16) ? 0 : 16;
    const int n0    = wave * 32 + l15;
    const int n1    = n0 + 16;
    const _Float16* Bp0 = Wf16 + (long)n0 * KTOT + kbb;  // state block: k in [0,256)
    const _Float16* Bp1 = Wf16 + (long)n1 * KTOT + kbb;
    const int mrow = (lane < 16) ? 0 : 8;
    const int nb0  = wave * 32;
    const int nb1  = nb0 + 16;
    const int rr   = wave * 2 + ((lane >= 16) ? 1 : 0); // row for gates/LN
    const float* srowf = &sfl[rr * 256];
    const float* urow  = &su[rr * 260];

    // ---- hoist W_s B-fragments (invariant over t): 16 x v16h in registers
    v16h bf0[8], bf1[8];
#pragma unroll
    for (int kk = 0; kk < 8; ++kk) {
        bf0[kk] = *(const v16h*)(Bp0 + kk * 32);
        bf1[kk] = *(const v16h*)(Bp1 + kk * 32);
    }

    for (int t = 0; t < T_; ++t) {
        // ---- u = W_s @ state (WMMA over K=256, 8 chunks, B in registers)
        v8f acc0 = {}, acc1 = {};
        const _Float16* Arow = &s16[l15 * 264];
#pragma unroll
        for (int kk = 0; kk < 8; ++kk) {
            v16h a = load_a_frag(Arow, kk * 32 + khalf);
            acc0 = wmma_f16(a, bf0[kk], acc0);
            acc1 = wmma_f16(a, bf1[kk], acc1);
        }

        // ---- gates on OLD state (half-wave: lanes 0-15 row 2w, 16-31 row 2w+1)
        float dk = 0.f, dq = 0.f;
#pragma unroll
        for (int j = 0; j < 16; ++j) {
            const int c = l15 + 16 * j;
            const float s = srowf[c];
            dk += s * W_sk[c];
            dq += s * W_sq[c];
        }
#pragma unroll
        for (int off = 8; off; off >>= 1) {
            dk += __shfl_xor(dk, off, 16);
            dq += __shfl_xor(dq, off, 16);
        }
        const long idx = (long)(b0 + rr) * T_ + t;
        if (l15 == 0) {
            const float rk = sigmoidf_(dk + pre_sk[idx]);
            const float rq = sigmoidf_(dq + pre_sq[idx]);
            out_preds[idx] = alpha * rk + (1.f - alpha) * rq;
        }

        // ---- u += pre[b,t,:] ; spill to LDS for the cross-tile LayerNorm
#pragma unroll
        for (int v = 0; v < 8; ++v) {
            const long prow = ((long)(b0 + mrow + v) * T_ + t) * 256;
            su[(mrow + v) * 260 + nb0 + l15] = acc0[v] + pre[prow + nb0 + l15];
            su[(mrow + v) * 260 + nb1 + l15] = acc1[v] + pre[prow + nb1 + l15];
        }
        // hide HBM latency of next step's pre tile (global_prefetch_b8)
        if (t + 1 < T_) {
            __builtin_prefetch(pre + ((long)(b0 + mrow) * T_ + (t + 1)) * 256 + nb0 + l15, 0, 3);
        }
        __syncthreads();

        // ---- LayerNorm + tanh for row rr (16 lanes, 16 elems each)
        float s1 = 0.f, s2 = 0.f;
#pragma unroll
        for (int j = 0; j < 16; ++j) {
            const float u = urow[l15 + 16 * j];
            s1 += u;
            s2 += u * u;
        }
#pragma unroll
        for (int off = 8; off; off >>= 1) {
            s1 += __shfl_xor(s1, off, 16);
            s2 += __shfl_xor(s2, off, 16);
        }
        const float mu   = s1 * (1.f / 256.f);
        const float var  = s2 * (1.f / 256.f) - mu * mu;
        const float rstd = __frsqrt_rn(var + 1e-5f);
        float* hsrow = out_hs + idx * 256;
#pragma unroll
        for (int j = 0; j < 16; ++j) {
            const int c = l15 + 16 * j;
            const float u  = urow[c];
            const float nv = tanhf((u - mu) * rstd * ln_g[c] + ln_b[c]);
            hsrow[c]          = nv;
            sfl[rr * 256 + c] = nv;
            s16[rr * 264 + c] = (_Float16)nv;
        }
        __syncthreads();
    }
}

// ---------------------------------------------------------------------------
extern "C" void kernel_launch(void* const* d_in, const int* in_sizes, int n_in,
                              void* d_out, int out_size, void* d_ws, size_t ws_size,
                              hipStream_t stream) {
    const int*   qids       = (const int*)  d_in[0];
    const int*   kcids      = (const int*)  d_in[1];
    const int*   resp       = (const int*)  d_in[2];
    const float* q_table    = (const float*)d_in[3];
    const float* kc_table   = (const float*)d_in[4];
    const float* resp_table = (const float*)d_in[5];
    const float* W_sk       = (const float*)d_in[6];
    const float* b_sk       = (const float*)d_in[7];
    const float* W_sq       = (const float*)d_in[8];
    const float* b_sq       = (const float*)d_in[9];
    const float* W_su       = (const float*)d_in[10];
    const float* b_su       = (const float*)d_in[11];
    const float* ln_g       = (const float*)d_in[12];
    const float* ln_b       = (const float*)d_in[13];
    const float* init_state = (const float*)d_in[14];
    const float* alpha_logit= (const float*)d_in[15];

    char* ws = (char*)d_ws;
    _Float16* Wf16 = (_Float16*)ws;                              // 786,432 B
    float* pre     = (float*)(ws + 786432);                      // 131,072,000 B
    float* pre_sk  = (float*)(ws + 786432 + (size_t)BT_ * 256 * 4);
    float* pre_sq  = pre_sk + BT_;

    float* out_preds = (float*)d_out;       // [B,T]
    float* out_hs    = out_preds + BT_;     // [B,T,SD]

    prep_w16<<<(SD_ * KTOT) / 256, 256, 0, stream>>>(W_su, Wf16);
    precompute_gemm<<<BT_ / MROWS, 256, MROWS * LDA * sizeof(_Float16), stream>>>(
        qids, kcids, resp, q_table, kc_table, resp_table, Wf16, b_su, pre);
    precompute_scalars<<<BT_ / 8, 256, 0, stream>>>(qids, kcids, q_table, kc_table,
                                                    W_sk, b_sk, W_sq, b_sq,
                                                    pre_sk, pre_sq);
    recurrence<<<B_ / 16, 256, 0, stream>>>(Wf16, pre, pre_sk, pre_sq,
                                            W_sk, W_sq, ln_g, ln_b,
                                            init_state, alpha_logit,
                                            out_preds, out_hs);
}
